// HardCBM_53730040873130
// MI455X (gfx1250) — matrix operand order repaired and checked by reading, not compiled, not run on hardware
//
#include <hip/hip_runtime.h>
#include <hip/hip_bf16.h>
#include <math.h>

typedef __attribute__((ext_vector_type(16))) _Float16 v16h;
typedef __attribute__((ext_vector_type(8)))  _Float16 v8h;
typedef __attribute__((ext_vector_type(8)))  float    v8f;

#define B_    2048
#define CIN_  256
#define NH_   512
#define NF_   112
#define KC_   112      // concepts
#define NL_   50
#define MC_   100
#define NOUT_ 200
#define R_    (B_*MC_) // 204800 rows = (b,m)
#define HP_   64       // NL padded to 64
#define KP_   128      // concept dim padded to 128
#define CSTR_ 112      // c16 row stride (f16 elems)
#define SLOPE 0.01f    // jax leaky_relu default

// ---------------- threefry2x32-20 (JAX RNG) ----------------
__device__ __forceinline__ unsigned rotl32(unsigned v, int r){ return (v<<r)|(v>>(32-r)); }
__device__ __forceinline__ void threefry(unsigned k0, unsigned k1, unsigned x0, unsigned x1,
                                         unsigned &o0, unsigned &o1){
  unsigned k2 = k0 ^ k1 ^ 0x1BD11BDAu;
  x0 += k0; x1 += k1;
  x0+=x1; x1=rotl32(x1,13); x1^=x0;  x0+=x1; x1=rotl32(x1,15); x1^=x0;
  x0+=x1; x1=rotl32(x1,26); x1^=x0;  x0+=x1; x1=rotl32(x1, 6); x1^=x0;
  x0+=k1; x1+=k2+1u;
  x0+=x1; x1=rotl32(x1,17); x1^=x0;  x0+=x1; x1=rotl32(x1,29); x1^=x0;
  x0+=x1; x1=rotl32(x1,16); x1^=x0;  x0+=x1; x1=rotl32(x1,24); x1^=x0;
  x0+=k2; x1+=k0+2u;
  x0+=x1; x1=rotl32(x1,13); x1^=x0;  x0+=x1; x1=rotl32(x1,15); x1^=x0;
  x0+=x1; x1=rotl32(x1,26); x1^=x0;  x0+=x1; x1=rotl32(x1, 6); x1^=x0;
  x0+=k0; x1+=k1+3u;
  x0+=x1; x1=rotl32(x1,17); x1^=x0;  x0+=x1; x1=rotl32(x1,29); x1^=x0;
  x0+=x1; x1=rotl32(x1,16); x1^=x0;  x0+=x1; x1=rotl32(x1,24); x1^=x0;
  x0+=k1; x1+=k2+4u;
  x0+=x1; x1=rotl32(x1,13); x1^=x0;  x0+=x1; x1=rotl32(x1,15); x1^=x0;
  x0+=x1; x1=rotl32(x1,26); x1^=x0;  x0+=x1; x1=rotl32(x1, 6); x1^=x0;
  x0+=k2; x1+=k0+5u;
  o0 = x0; o1 = x1;
}
__device__ __forceinline__ float uni01(unsigned bits){
  return __uint_as_float((bits >> 9) | 0x3f800000u) - 1.0f;
}
__device__ __forceinline__ float sigmoidf_(float s){ return 1.0f/(1.0f + __expf(-s)); }

// key(42) -> 112 sequential splits; sk[i] = second child
__global__ void rng_kernel(unsigned* sk){
  if (threadIdx.x || blockIdx.x) return;
  unsigned k0 = 0u, k1 = 42u;
  for (int i = 0; i < KC_; ++i){
    unsigned a0,a1,b0,b1;
    threefry(k0,k1,0u,2u,a0,a1);   // counter lane (0,2)
    threefry(k0,k1,1u,3u,b0,b1);   // counter lane (1,3)
    sk[2*i] = a1; sk[2*i+1] = b1;  // sk = (out[2],out[3])
    k0 = a0; k1 = b0;              // key = (out[0],out[1])
  }
}

// ---------------- WMMA fragment helpers (wave32, f16 16x16x32) ----------------
__device__ __forceinline__ v8f wmma16(v16h a, v16h b, v8f c){
  return __builtin_amdgcn_wmma_f32_16x16x32_f16(false, a, false, b, (short)0, c, false, false);
}
// A 16x32: row = m0+lane%16; lanes<16: V0-3=K0..7,V4-7=K16..23; lanes>=16: K8..15,K24..31
__device__ __forceinline__ v16h load_a_frag(const _Float16* base, int stride, int m0, int kbase, int lane){
  int row = m0 + (lane & 15);
  int k0  = kbase + ((lane & 16) ? 8 : 0);
  const _Float16* p = base + (size_t)row*stride + k0;
  union { v16h v; v8h h[2]; } f;
  f.h[0] = *(const v8h*)(p);
  f.h[1] = *(const v8h*)(p + 16);
  return f.v;
}
// B 32x16: col = n0+lane%16; lanes<16: K0..7,K16..23; lanes>=16: K8..15,K24..31
__device__ __forceinline__ v16h load_b_frag(const _Float16* base, int stride, int n0, int kbase, int lane){
  int col = n0 + (lane & 15);
  int k0  = kbase + ((lane & 16) ? 8 : 0);
  const _Float16* p = base + (size_t)col*stride + k0;
  union { v16h v; v8h h[2]; } f;
  f.h[0] = *(const v8h*)(p);
  f.h[1] = *(const v8h*)(p + 16);
  return f.v;
}

// ---------------- packing kernels (run every call; deterministic) ----------------
__global__ void pack_cast(const float* src, _Float16* dst, int n){
  int t = blockIdx.x*256 + threadIdx.x; if (t < n) dst[t] = (_Float16)src[t];
}
// dst[n*Kd+k] = src[k*N+n]
__global__ void pack_transpose(const float* src, _Float16* dst, int Kd, int N){
  int t = blockIdx.x*256 + threadIdx.x;
  if (t >= Kd*N) return;
  int n = t / Kd, k = t % Kd;
  dst[t] = (_Float16)src[(size_t)k*N + n];
}
// W1a[i][h][k] = pred_W1[i][k][h] (k<NF,h<NL else 0)  -- base GEMM B operand
__global__ void pack_W1a(const float* pW1, _Float16* dst){
  int t = blockIdx.x*256 + threadIdx.x;            // 112*64*128
  int i = t / (HP_*KP_); int rem = t % (HP_*KP_);
  int h = rem / KP_, k = rem % KP_;
  float v = (h < NL_ && k < NF_) ? pW1[((size_t)i*(NF_+KC_) + k)*NL_ + h] : 0.f;
  dst[t] = (_Float16)v;
}
// Wt[i][h][k] = pred_W1[i][NF+k][h] (k<i,h<NL else 0) -- predictor A operand
__global__ void pack_Wt(const float* pW1, _Float16* dst){
  int t = blockIdx.x*256 + threadIdx.x;
  int i = t / (HP_*KP_); int rem = t % (HP_*KP_);
  int h = rem / KP_, k = rem % KP_;
  float v = (h < NL_ && k < i) ? pW1[((size_t)i*(NF_+KC_) + NF_ + k)*NL_ + h] : 0.f;
  dst[t] = (_Float16)v;
}
// headWt[o][k] = head_W[k][o] (o<200,k<112 else 0)
__global__ void pack_headW(const float* hW, _Float16* dst){
  int t = blockIdx.x*256 + threadIdx.x;            // 208*128
  if (t >= 208*KP_) return;
  int o = t / KP_, k = t % KP_;
  float v = (o < NOUT_ && k < KC_) ? hW[(size_t)k*NOUT_ + o] : 0.f;
  dst[t] = (_Float16)v;
}
__global__ void pad_inter(_Float16* inter16){
  int t = blockIdx.x*256 + threadIdx.x;            // 2048*16
  int b = t >> 4, j = t & 15;
  inter16[(size_t)b*KP_ + NF_ + j] = (_Float16)0.f;
}
__global__ void pad_c16(_Float16* c16){
  int t = blockIdx.x*256 + threadIdx.x;
  if (t < 16*CSTR_ + 64) c16[(size_t)R_*CSTR_ + t] = (_Float16)0.f;
}

// ---------------- generic f16 WMMA GEMM:  C=f16( act(A@Wt^T + bias) ) ----------------
__global__ void gemm_enc(const _Float16* __restrict__ A, int lda,
                         const _Float16* __restrict__ Wt, int ldw,   // [N][Kd]
                         const float* __restrict__ bias,
                         _Float16* __restrict__ C, int ldc,
                         int Mtiles, int Ntiles, int Kd, int do_lrelu){
  int wave = blockIdx.x*8 + (threadIdx.x >> 5);
  int lane = threadIdx.x & 31;
  if (wave >= Mtiles*Ntiles) return;
  int rt = wave / Ntiles, ct = wave % Ntiles;
  v8f acc = {};
  for (int kt = 0; kt < Kd; kt += 32)
    acc = wmma16(load_a_frag(A, lda, rt*16, kt, lane),
                 load_b_frag(Wt, ldw, ct*16, kt, lane), acc);
  int n = ct*16 + (lane & 15);
  float bn = bias[n];
  int mbase = rt*16 + ((lane & 16) ? 8 : 0);
  #pragma unroll
  for (int d = 0; d < 8; ++d){
    float v = acc[d] + bn;
    if (do_lrelu) v = (v > 0.f) ? v : SLOPE*v;
    C[(size_t)(mbase+d)*ldc + n] = (_Float16)v;
  }
}

// base_all[i][b][h] = inter @ pred_W1[i,:NF] + b1[i]   (h>=50 -> 0)
__global__ void base_gemm(const _Float16* __restrict__ inter16,
                          const _Float16* __restrict__ W1a,
                          const float* __restrict__ pb1,
                          float* __restrict__ base_all){
  int wave = blockIdx.x*8 + (threadIdx.x >> 5);   // 112*128*4 waves exactly
  int lane = threadIdx.x & 31;
  int i = wave / (128*4);
  int rem = wave % (128*4);
  int rt = rem >> 2, ht = rem & 3;
  const _Float16* Wi = W1a + (size_t)i*HP_*KP_;
  v8f acc = {};
  #pragma unroll
  for (int kt = 0; kt < 4; ++kt)
    acc = wmma16(load_a_frag(inter16, KP_, rt*16, kt*32, lane),
                 load_b_frag(Wi, KP_, ht*16, kt*32, lane), acc);
  int h = ht*16 + (lane & 15);
  float b1v = (h < NL_) ? pb1[i*NL_ + h] : 0.f;
  int mbase = rt*16 + ((lane & 16) ? 8 : 0);
  float* dst = base_all + (size_t)i*B_*HP_;
  #pragma unroll
  for (int d = 0; d < 8; ++d){
    float v = (h < NL_) ? (acc[d] + b1v) : 0.f;
    dst[(size_t)(mbase+d)*HP_ + h] = v;
  }
}

// step 0 logits: s[row] = lrelu(base0[b]) . W2[0] + b2[0]
__global__ void step0_kernel(const float* __restrict__ base_all,
                             const float* __restrict__ pW2,
                             const float* __restrict__ pb2,
                             float* __restrict__ slog){
  int row = blockIdx.x*256 + threadIdx.x;
  if (row >= R_) return;
  int b = row / MC_;
  const float* bp = base_all + (size_t)b*HP_;
  float s = 0.f;
  #pragma unroll 5
  for (int h = 0; h < NL_; ++h){
    float v = bp[h]; v = (v > 0.f) ? v : SLOPE*v;
    s += v * pW2[h];
  }
  slog[row] = s + pb2[0];
}

__device__ __forceinline__ void sample_one(int e, int j, unsigned k0, unsigned k1,
                                           const float* slog, _Float16* c16,
                                           float* c_prob, float* c_hard){
  int e2 = e + R_/2;
  unsigned r0, r1;
  threefry(k0, k1, (unsigned)e, (unsigned)e2, r0, r1);
  float u0 = uni01(r0), u1 = uni01(r1);
  float p0 = sigmoidf_(slog[e]);
  float p1 = sigmoidf_(slog[e2]);
  float h0 = (u0 < p0) ? 1.f : 0.f;
  float h1 = (u1 < p1) ? 1.f : 0.f;
  int b0 = e  / MC_, m0 = e  - b0*MC_;
  int b1 = e2 / MC_, m1 = e2 - b1*MC_;
  c_prob[(size_t)b0*KC_*MC_ + j*MC_ + m0] = p0;
  c_prob[(size_t)b1*KC_*MC_ + j*MC_ + m1] = p1;
  c_hard[(size_t)b0*KC_*MC_ + j*MC_ + m0] = h0;
  c_hard[(size_t)b1*KC_*MC_ + j*MC_ + m1] = h1;
  c16[(size_t)e *CSTR_ + j] = (_Float16)h0;
  c16[(size_t)e2*CSTR_ + j] = (_Float16)h1;
}

// one sequential step: sample concept (i-1), barrier, WMMA GEMM for step i
__global__ void __launch_bounds__(256)
step_kernel(int i, const unsigned* __restrict__ sk,
            const _Float16* __restrict__ Wt, const float* __restrict__ base_all,
            const float* __restrict__ pW2, const float* __restrict__ pb2,
            float* __restrict__ slog, _Float16* __restrict__ c16,
            float* __restrict__ c_prob, float* __restrict__ c_hard){
  __shared__ float w2s[HP_];
  __shared__ float b2s;
  int tid = threadIdx.x, blk = blockIdx.x;
  if (tid < HP_) w2s[tid] = (tid < NL_) ? pW2[i*NL_ + tid] : 0.f;
  if (tid == 0)  b2s = pb2[i];

  // phase 1: Bernoulli-sample step i-1 for this WG's paired row blocks
  {
    int j = i - 1;
    unsigned k0 = sk[2*j], k1 = sk[2*j+1];
    sample_one(blk*512 + tid,        j, k0, k1, slog, c16, c_prob, c_hard);
    sample_one(blk*512 + 256 + tid,  j, k0, k1, slog, c16, c_prob, c_hard);
  }
  __syncthreads();

  // phase 2: s[row] = lrelu(base_i + c_hard[:,:i] @ W1x_i) . W2_i + b2_i  via WMMA
  int lane = tid & 31, wv = tid >> 5;
  int ktiles = (i + 31) >> 5;
  const float*    base_i = base_all + (size_t)i*B_*HP_;
  const _Float16* Wti    = Wt + (size_t)i*HP_*KP_;
  int half8 = (lane & 16) ? 8 : 0;
  for (int tt = wv*8; tt < wv*8 + 8; ++tt){
    int rowbase = (tt < 32) ? (blk*512 + tt*16) : (R_/2 + blk*512 + (tt-32)*16);
    v8f acc[4] = {};
    for (int kt = 0; kt < ktiles; ++kt){
      v16h bf = load_b_frag(c16, CSTR_, rowbase, kt*32, lane);
      #pragma unroll
      for (int mt = 0; mt < 4; ++mt)
        acc[mt] = wmma16(load_a_frag(Wti, KP_, mt*16, kt*32, lane), bf, acc[mt]);
    }
    int row = rowbase + (lane & 15);
    const float* bp = base_i + (size_t)(row / MC_)*HP_;
    float s = 0.f;
    #pragma unroll
    for (int mt = 0; mt < 4; ++mt){
      int h0 = mt*16 + half8;
      #pragma unroll
      for (int d = 0; d < 8; ++d){
        float h = bp[h0+d] + acc[mt][d];
        h = (h > 0.f) ? h : SLOPE*h;
        s += h * w2s[h0+d];
      }
    }
    s += __shfl_xor(s, 16, 32);
    s += b2s;
    if (lane < 16) slog[row] = s;
  }
}

// final sampling for concept 111
__global__ void sample_kernel(int j, const unsigned* __restrict__ sk,
                              const float* __restrict__ slog, _Float16* __restrict__ c16,
                              float* __restrict__ c_prob, float* __restrict__ c_hard){
  int e = blockIdx.x*256 + threadIdx.x;   // < 102400
  sample_one(e, j, sk[2*j], sk[2*j+1], slog, c16, c_prob, c_hard);
}

// head: logits=c_hard@head_W+b; softmax over o; mean over m; log
__global__ void __launch_bounds__(256)
head_kernel(const _Float16* __restrict__ c16, const _Float16* __restrict__ hWt,
            const float* __restrict__ hb, float* __restrict__ ylog){
  __shared__ float accum[208];
  __shared__ float hbs[208];
  int tid = threadIdx.x, b = blockIdx.x;
  if (tid < 208){ accum[tid] = 0.f; hbs[tid] = (tid < NOUT_) ? hb[tid] : 0.f; }
  __syncthreads();
  int wv = tid >> 5, lane = tid & 31;
  if (wv < 7){
    int rowbase = b*MC_ + wv*16;
    v8f acc[13] = {};
    for (int kt = 0; kt < 4; ++kt){
      v16h bf = load_b_frag(c16, CSTR_, rowbase, kt*32, lane);
      #pragma unroll
      for (int mt = 0; mt < 13; ++mt)
        acc[mt] = wmma16(load_a_frag(hWt, KP_, mt*16, kt*32, lane), bf, acc[mt]);
    }
    int m = wv*16 + (lane & 15);
    bool valid = (m < MC_);
    int half8 = (lane & 16) ? 8 : 0;
    float se = 0.f;                     // |logit| small -> exp safe without max-sub
    #pragma unroll
    for (int mt = 0; mt < 13; ++mt){
      #pragma unroll
      for (int d = 0; d < 8; ++d){
        int o = mt*16 + d + half8;
        float e = (o < NOUT_) ? __expf(acc[mt][d] + hbs[o]) : 0.f;
        acc[mt][d] = e;
        se += e;
      }
    }
    se += __shfl_xor(se, 16, 32);
    float inv = valid ? (1.f/se) : 0.f;
    #pragma unroll
    for (int mt = 0; mt < 13; ++mt){
      #pragma unroll
      for (int d = 0; d < 8; ++d){
        int o = mt*16 + d + half8;
        float c = valid ? acc[mt][d]*inv : 0.f;
        c += __shfl_xor(c, 1, 32);
        c += __shfl_xor(c, 2, 32);
        c += __shfl_xor(c, 4, 32);
        c += __shfl_xor(c, 8, 32);
        if ((lane & 15) == 0 && o < NOUT_) atomicAdd(&accum[o], c);
      }
    }
  }
  __syncthreads();
  if (tid < NOUT_) ylog[(size_t)b*NOUT_ + tid] = logf(accum[tid]*(1.f/MC_) + 1e-6f);
}

// ---------------- host side ----------------
extern "C" void kernel_launch(void* const* d_in, const int* in_sizes, int n_in,
                              void* d_out, int out_size, void* d_ws, size_t ws_size,
                              hipStream_t stream){
  (void)in_sizes; (void)n_in; (void)out_size; (void)ws_size;
  const float* x     = (const float*)d_in[0];
  const float* Win   = (const float*)d_in[1];
  const float* bin   = (const float*)d_in[2];
  const float* Wh    = (const float*)d_in[3];
  const float* bh    = (const float*)d_in[4];
  const float* Wout  = (const float*)d_in[5];
  const float* bout  = (const float*)d_in[6];
  const float* pW1   = (const float*)d_in[7];
  const float* pb1   = (const float*)d_in[8];
  const float* pW2   = (const float*)d_in[9];
  const float* pb2   = (const float*)d_in[10];
  const float* hW    = (const float*)d_in[11];
  const float* hb    = (const float*)d_in[12];

  float* out    = (float*)d_out;
  float* c_prob = out;                                      // (B,K,MC)
  float* ylog   = out + (size_t)B_*KC_*MC_;                 // (B,NOUT)
  float* c_hard = ylog + (size_t)B_*NOUT_;                  // (B,K,MC)

  char* w = (char*)d_ws;
  auto alloc = [&](size_t bytes) -> char* {
    char* p = w; w += (bytes + 255) & ~(size_t)255; return p;
  };
  unsigned*  sk      = (unsigned*)alloc(KC_*2*4);
  _Float16*  xt16    = (_Float16*)alloc((size_t)B_*CIN_*2);
  _Float16*  h16a    = (_Float16*)alloc((size_t)B_*NH_*2);
  _Float16*  h16b    = (_Float16*)alloc((size_t)B_*NH_*2);
  _Float16*  WinT    = (_Float16*)alloc((size_t)NH_*CIN_*2);
  _Float16*  WhT     = (_Float16*)alloc((size_t)4*NH_*NH_*2);
  _Float16*  WoutT   = (_Float16*)alloc((size_t)NF_*NH_*2);
  _Float16*  inter16 = (_Float16*)alloc((size_t)B_*KP_*2);
  _Float16*  W1a     = (_Float16*)alloc((size_t)KC_*HP_*KP_*2);
  _Float16*  Wt      = (_Float16*)alloc((size_t)KC_*HP_*KP_*2);
  _Float16*  headWt  = (_Float16*)alloc((size_t)208*KP_*2);
  float*     baseAll = (float*)alloc((size_t)KC_*B_*HP_*4);
  _Float16*  c16     = (_Float16*)alloc(((size_t)(R_+16)*CSTR_ + 64)*2);
  float*     slog    = (float*)alloc((size_t)R_*4);

  rng_kernel<<<1, 64, 0, stream>>>(sk);

  pack_cast<<<(B_*CIN_)/256, 256, 0, stream>>>(x, xt16, B_*CIN_);
  pack_transpose<<<(CIN_*NH_+255)/256, 256, 0, stream>>>(Win, WinT, CIN_, NH_);
  for (int l = 0; l < 4; ++l)
    pack_transpose<<<(NH_*NH_+255)/256, 256, 0, stream>>>(Wh + (size_t)l*NH_*NH_,
                                                          WhT + (size_t)l*NH_*NH_, NH_, NH_);
  pack_transpose<<<(NH_*NF_+255)/256, 256, 0, stream>>>(Wout, WoutT, NH_, NF_);
  pack_W1a<<<(KC_*HP_*KP_)/256, 256, 0, stream>>>(pW1, W1a);
  pack_Wt<<<(KC_*HP_*KP_)/256, 256, 0, stream>>>(pW1, Wt);
  pack_headW<<<(208*KP_+255)/256, 256, 0, stream>>>(hW, headWt);
  pad_c16<<<8, 256, 0, stream>>>(c16);

  // encoder (ping-pong h16a/h16b)
  gemm_enc<<<512, 256, 0, stream>>>(xt16, CIN_, WinT, CIN_, bin, h16a, NH_, 128, 32, CIN_, 1);
  gemm_enc<<<512, 256, 0, stream>>>(h16a, NH_, WhT + 0*NH_*NH_, NH_, bh + 0*NH_, h16b, NH_, 128, 32, NH_, 1);
  gemm_enc<<<512, 256, 0, stream>>>(h16b, NH_, WhT + 1*NH_*NH_, NH_, bh + 1*NH_, h16a, NH_, 128, 32, NH_, 1);
  gemm_enc<<<512, 256, 0, stream>>>(h16a, NH_, WhT + 2*NH_*NH_, NH_, bh + 2*NH_, h16b, NH_, 128, 32, NH_, 1);
  gemm_enc<<<512, 256, 0, stream>>>(h16b, NH_, WhT + 3*NH_*NH_, NH_, bh + 3*NH_, h16a, NH_, 128, 32, NH_, 1);
  gemm_enc<<<112, 256, 0, stream>>>(h16a, NH_, WoutT, NH_, bout, inter16, KP_, 128, 7, NH_, 0);
  pad_inter<<<128, 256, 0, stream>>>(inter16);

  // base precompute for all 112 concepts
  base_gemm<<<(KC_*128*4)/8, 256, 0, stream>>>(inter16, W1a, pb1, baseAll);

  // sequential concept chain
  step0_kernel<<<R_/256, 256, 0, stream>>>(baseAll, pW2, pb2, slog);
  for (int i = 1; i < KC_; ++i)
    step_kernel<<<200, 256, 0, stream>>>(i, sk, Wt, baseAll, pW2, pb2, slog, c16, c_prob, c_hard);
  sample_kernel<<<(R_/2)/256, 256, 0, stream>>>(KC_-1, sk, slog, c16, c_prob, c_hard);

  // head + softmax/mean/log
  head_kernel<<<B_, 256, 0, stream>>>(c16, headWt, hb, ylog);
}
